// get_model_87660282511916
// MI455X (gfx1250) — matrix operand order repaired and checked by reading, not compiled, not run on hardware
//
#include <hip/hip_runtime.h>
#include <cstdint>
#include <cstddef>

// ============================================================================
// PointNet++-style model forward for MI455X (gfx1250, wave32, WMMA).
//
// Data layout: ALL activations are point-major f16 [cols][Cpad], Cpad =
// round_up(C,32), padding lanes kept zero. Consequences:
//   - WMMA B fragment = 16 contiguous f16 per lane  -> 2x global_load_b128
//   - WMMA A fragment from device-prepacked weights -> 2x global_load_b128
//   - WMMA D written transposed (point-major)       -> 1x global_store_b128
//   - no LDS, no __syncthreads, no bounds checks in the GEMM K-loop
// Conv bias + BN folded into per-channel scale/shift (zeroed on pad channels
// so padded outputs are exactly 0 == next layer's K padding). All column
// counts in this graph are multiples of 64, so no N-edge handling.
// sa13/fp13 are dead in the reference graph (pruned); FPS(2048) over the
// same point set is a prefix of FPS(4096) (reused).
// ============================================================================

typedef _Float16 h16;
typedef __attribute__((ext_vector_type(16))) _Float16 v16h;
typedef __attribute__((ext_vector_type(8)))  float    v8f;

#define TPB 256
static inline unsigned gr(long long n){ return (unsigned)((n + TPB - 1) / TPB); }
static inline int pad32(int c){ return (c + 31) & ~31; }

// ---------------------------------------------------------------------------
// Weight packing: f32 [M][K] -> u32 stream in WMMA A-fragment order:
//   tile (tm, tk) -> 32 lanes -> 8 dwords ; dword d of lane l holds
//   (m = tm*16 + (l&15), k = tk*32 + (l>>4)*16 + d*2) and k+1.
// ---------------------------------------------------------------------------
__global__ void k_packA(const float* __restrict__ w, unsigned* __restrict__ out,
                        int M, int K, int Mpad, int Kpad){
  long long total = (long long)(Mpad / 16) * (Kpad / 32) * 256;
  long long t = (long long)blockIdx.x * blockDim.x + threadIdx.x;
  if (t >= total) return;
  int d    = (int)(t & 7);
  int lane = (int)((t >> 3) & 31);
  long long tile = t >> 8;
  int nk = Kpad >> 5;
  int tk = (int)(tile % nk);
  int tm = (int)(tile / nk);
  int m = tm * 16 + (lane & 15);
  int k = tk * 32 + ((lane >> 4) * 16) + d * 2;
  union { h16 h[2]; unsigned u; } p;
  p.h[0] = (m < M && k     < K) ? (h16)w[(size_t)m * K + k]     : (h16)0.f;
  p.h[1] = (m < M && k + 1 < K) ? (h16)w[(size_t)m * K + k + 1] : (h16)0.f;
  out[t] = p.u;
}

// fold conv-bias + BN into scale/shift over padded channels (pads -> 0)
__global__ void k_fold_bn(const float* __restrict__ cb, const float* __restrict__ g,
                          const float* __restrict__ bb, const float* __restrict__ m,
                          const float* __restrict__ v, float* __restrict__ sc,
                          float* __restrict__ sh, int C, int Cp){
  int c = blockIdx.x * blockDim.x + threadIdx.x;
  if (c >= Cp) return;
  if (c < C){ float s = g[c] * rsqrtf(v[c] + 1e-5f); sc[c] = s; sh[c] = (cb[c] - m[c]) * s + bb[c]; }
  else { sc[c] = 0.f; sh[c] = 0.f; }
}
__global__ void k_fold_bias(const float* __restrict__ cb, float* __restrict__ sc,
                            float* __restrict__ sh, int C, int Cp){
  int c = blockIdx.x * blockDim.x + threadIdx.x;
  if (c >= Cp) return;
  sc[c] = (c < C) ? 1.f : 0.f;
  sh[c] = (c < C) ? cb[c] : 0.f;
}

// ---------------------------------------------------------------------------
// WMMA GEMM, fully regular: out_T[n][Mpad] = epi( A[M,K] * B_T[n][Kpad]^T )
// block = 8 waves (2 along M, 4 along N) -> 32 x 64 tile, K-step 32.
// ---------------------------------------------------------------------------
__launch_bounds__(256)
__global__ void k_gemm(const uint4* __restrict__ Apk, const h16* __restrict__ Bt,
                       const float* __restrict__ scale, const float* __restrict__ shift,
                       h16* __restrict__ Ct, int Mpad, int Kpad, int relu)
{
  const int tid  = threadIdx.x;
  const int lane = tid & 31;
  const int wave = tid >> 5;
  const int wm   = wave >> 2;            // 0..1
  const int wn   = wave & 3;             // 0..3
  const int nk   = Kpad >> 5;
  const int tm   = blockIdx.y * 2 + wm;

  const uint4* __restrict__ ap = Apk + (size_t)tm * nk * 64 + (size_t)lane * 2;
  const long long n = (long long)blockIdx.x * 64 + wn * 16 + (lane & 15);
  const h16* __restrict__ bp = Bt + (size_t)n * Kpad + ((lane >> 4) * 16);

  v8f acc = {0.f, 0.f, 0.f, 0.f, 0.f, 0.f, 0.f, 0.f};

  for (int kt = 0; kt < nk; ++kt) {
    union { uint4 q[2]; v16h v; } ua, ub;
    ua.q[0] = ap[0];
    ua.q[1] = ap[1];
    ub.q[0] = *(const uint4*)(bp);
    ub.q[1] = *(const uint4*)(bp + 8);
    if (kt + 1 < nk) {
      __builtin_prefetch(ap + 64, 0, 0);
      __builtin_prefetch(bp + 32, 0, 0);
    }
    acc = __builtin_amdgcn_wmma_f32_16x16x32_f16(false, ua.v, false, ub.v,
                                                 (short)0, acc, false, false);
    ap += 64;
    bp += 32;
  }

  // lane holds M rows mb..mb+7 at column n (contiguous in point-major out)
  const int mb = tm * 16 + (lane >> 4) * 8;
  union { h16 h[8]; uint4 q; } ov;
#pragma unroll
  for (int r = 0; r < 8; ++r) {
    float x = acc[r] * scale[mb + r] + shift[mb + r];
    if (relu) x = x > 0.f ? x : 0.f;
    ov.h[r] = (h16)x;
  }
  *(uint4*)(Ct + (size_t)n * Mpad + mb) = ov.q;
}

// GEMV for the STN FC layers (single column); weights read as f32 directly
__global__ void k_gemv(const float* __restrict__ w, const h16* __restrict__ x,
                       const float* __restrict__ sc, const float* __restrict__ sh,
                       h16* yh, float* yf, int M, int K, int relu){
  int m = blockIdx.x * blockDim.x + threadIdx.x;
  if (m >= M) return;
  float acc = 0.f;
  for (int k = 0; k < K; ++k) acc += w[(size_t)m * K + k] * (float)x[k];
  float y = acc * sc[m] + sh[m];
  if (relu && y < 0.f) y = 0.f;
  if (yh) yh[m] = (h16)y;
  if (yf) yf[m] = y;
}

// column-wise max over rows of point-major [rows][Kp] -> vec[Kp]
__global__ void k_colmax(const h16* __restrict__ in, h16* __restrict__ vec,
                         long long rows, int Kp){
  int c = blockIdx.x * blockDim.x + threadIdx.x;
  if (c >= Kp) return;
  float m = -3.0e38f;
  for (long long r = 0; r < rows; ++r) {
    float v = (float)in[(size_t)r * Kp + c];
    if (v > m) m = v;
  }
  vec[c] = (h16)m;
}

// max over sample dim: in[(p*S+s)][Cp] -> out[p][Cp]
__global__ void k_maxpool(const h16* __restrict__ in, h16* __restrict__ out,
                          int P, int S, int Cp){
  long long t = (long long)blockIdx.x * blockDim.x + threadIdx.x;
  if (t >= (long long)P * Cp) return;
  int p = (int)(t / Cp), c = (int)(t % Cp);
  float m = -3.0e38f;
  for (int s = 0; s < S; ++s) {
    float v = (float)in[((size_t)p * S + s) * Cp + c];
    if (v > m) m = v;
  }
  out[(size_t)p * Cp + c] = (h16)m;
}

// xyz f32 [3][N] -> point-major f32 [N][3] and f16 rows [N][32] (pad 0)
__global__ void k_prep_xyz(const float* __restrict__ xyz, float* __restrict__ pm,
                           h16* __restrict__ xT, int n){
  int i = blockIdx.x * blockDim.x + threadIdx.x;
  if (i >= n) return;
  float x = xyz[i], y = xyz[n + i], z = xyz[2 * n + i];
  pm[(size_t)i * 3] = x; pm[(size_t)i * 3 + 1] = y; pm[(size_t)i * 3 + 2] = z;
  h16* r = xT + (size_t)i * 32;
  r[0] = (h16)x; r[1] = (h16)y; r[2] = (h16)z;
  for (int c = 3; c < 32; ++c) r[c] = (h16)0.f;
}

// e[n][d] = sum_c xyz[c][n] * T[c*3+d], rows [N][32]
__global__ void k_transform(const float* __restrict__ xyz, const float* __restrict__ T,
                            h16* __restrict__ eT, int n){
  int i = blockIdx.x * blockDim.x + threadIdx.x;
  if (i >= n) return;
  float x = xyz[i], y = xyz[n + i], z = xyz[2 * n + i];
  h16* r = eT + (size_t)i * 32;
  for (int d = 0; d < 3; ++d) r[d] = (h16)(x * T[d] + y * T[3 + d] + z * T[6 + d]);
  for (int c = 3; c < 32; ++c) r[c] = (h16)0.f;
}
__global__ void k_add_eye(float* T){
  if (threadIdx.x == 0) { T[0] += 1.f; T[4] += 1.f; T[8] += 1.f; }
}

// concat of two point-major sources (+ zero pad):  dst[n][c] over [rows][dKp]
__global__ void k_cat2(h16* __restrict__ dst, int dKp,
                       const h16* __restrict__ a, int aKp, int Ca,
                       const h16* __restrict__ b, int bKp, int Cb, long long rows){
  long long t = (long long)blockIdx.x * blockDim.x + threadIdx.x;
  if (t >= rows * dKp) return;
  long long nrow = t / dKp; int c = (int)(t % dKp);
  h16 v = (h16)0.f;
  if (c < Ca) v = a[(size_t)nrow * aKp + c];
  else if (b && c < Ca + Cb) v = b[(size_t)nrow * bKp + (c - Ca)];
  dst[t] = v;
}
// [broadcast vec | feature rows] concat (head global feature)
__global__ void k_vec_cat(h16* __restrict__ dst, int dKp,
                          const h16* __restrict__ vec, int Cv,
                          const h16* __restrict__ f, int fKp, int Cf, long long rows){
  long long t = (long long)blockIdx.x * blockDim.x + threadIdx.x;
  if (t >= rows * dKp) return;
  long long nrow = t / dKp; int c = (int)(t % dKp);
  h16 v = (h16)0.f;
  if (c < Cv) v = vec[c];
  else if (c < Cv + Cf) v = f[(size_t)nrow * fKp + (c - Cv)];
  dst[t] = v;
}
// [feature rows | broadcast vec] concat (fp03 / group_all propagate)
__global__ void k_featvec_cat(h16* __restrict__ dst, int dKp,
                              const h16* __restrict__ f, int fKp, int Cf,
                              const h16* __restrict__ vec, int Cv, long long rows){
  long long t = (long long)blockIdx.x * blockDim.x + threadIdx.x;
  if (t >= rows * dKp) return;
  long long nrow = t / dKp; int c = (int)(t % dKp);
  h16 v = (h16)0.f;
  if (c < Cf) v = f[(size_t)nrow * fKp + c];
  else if (c < Cf + Cv) v = vec[c - Cf];
  dst[t] = v;
}
// [xyz(from f32 point-major) | feature rows] concat (sa03 group_all input)
__global__ void k_pm_cat(h16* __restrict__ dst, int dKp,
                         const float* __restrict__ pm,
                         const h16* __restrict__ f, int fKp, int Cf, long long rows){
  long long t = (long long)blockIdx.x * blockDim.x + threadIdx.x;
  if (t >= rows * dKp) return;
  long long nrow = t / dKp; int c = (int)(t % dKp);
  h16 v = (h16)0.f;
  if (c < 3) v = (h16)pm[(size_t)nrow * 3 + c];
  else if (c < 3 + Cf) v = f[(size_t)nrow * fKp + (c - 3)];
  dst[t] = v;
}

// log-softmax over 14 logits; writes f32 seg (channel-major, model output) and
// builds l0p00 rows [xyz(3) | seg(14) | 0...] in one pass
__global__ void k_logsm14(const h16* __restrict__ lg, const h16* __restrict__ xT,
                          float* __restrict__ outf, h16* __restrict__ l0p00, int n){
  int i = blockIdx.x * blockDim.x + threadIdx.x;
  if (i >= n) return;
  float v[14]; float m = -3.0e38f;
  for (int c = 0; c < 14; ++c) { v[c] = (float)lg[(size_t)i * 32 + c]; if (v[c] > m) m = v[c]; }
  float s = 0.f;
  for (int c = 0; c < 14; ++c) s += expf(v[c] - m);
  float ls = logf(s);
  h16* r = l0p00 + (size_t)i * 32;
  r[0] = xT[(size_t)i * 32]; r[1] = xT[(size_t)i * 32 + 1]; r[2] = xT[(size_t)i * 32 + 2];
  for (int c = 0; c < 14; ++c) {
    float o = v[c] - m - ls;
    outf[(size_t)c * n + i] = o;
    r[3 + c] = (h16)o;
  }
  for (int c = 17; c < 32; ++c) r[c] = (h16)0.f;
}

// farthest point sampling: one resident block, LDS argmax reduction
__launch_bounds__(1024)
__global__ void k_fps(const float* __restrict__ pm, int n, int npoint,
                      int* __restrict__ idx, float* __restrict__ dist){
  __shared__ float sv[1024];
  __shared__ int   si[1024];
  int tid = threadIdx.x;
  for (int i = tid; i < n; i += 1024) dist[i] = 1e10f;
  __syncthreads();
  int far = 0;
  for (int it = 0; it < npoint; ++it) {
    if (tid == 0) idx[it] = far;
    float cx = pm[(size_t)far * 3], cy = pm[(size_t)far * 3 + 1], cz = pm[(size_t)far * 3 + 2];
    float bv = -1.f; int bi = 0;
    for (int i = tid; i < n; i += 1024) {
      float dx = pm[(size_t)i * 3] - cx, dy = pm[(size_t)i * 3 + 1] - cy, dz = pm[(size_t)i * 3 + 2] - cz;
      float d = dx * dx + dy * dy + dz * dz;
      float nd = fminf(dist[i], d);
      dist[i] = nd;
      if (nd > bv) { bv = nd; bi = i; }
    }
    sv[tid] = bv; si[tid] = bi; __syncthreads();
    for (int s = 512; s > 0; s >>= 1) {
      if (tid < s && sv[tid + s] > sv[tid]) { sv[tid] = sv[tid + s]; si[tid] = si[tid + s]; }
      __syncthreads();
    }
    far = si[0];
    __syncthreads();
  }
}

__global__ void k_gather_xyz(const float* __restrict__ pm, const int* __restrict__ idx,
                             int P, float* __restrict__ out){
  int p = blockIdx.x * blockDim.x + threadIdx.x;
  if (p >= P) return;
  int j = idx[p];
  out[(size_t)p * 3]     = pm[(size_t)j * 3];
  out[(size_t)p * 3 + 1] = pm[(size_t)j * 3 + 1];
  out[(size_t)p * 3 + 2] = pm[(size_t)j * 3 + 2];
}

// ball query: first S in-index-order neighbors within radius, padded with first
__global__ void k_ballquery(const float* __restrict__ pm, const float* __restrict__ qpm,
                            int n, int P, int S, float r2, int* __restrict__ idx){
  int p = blockIdx.x * blockDim.x + threadIdx.x;
  if (p >= P) return;
  float qx = qpm[(size_t)p * 3], qy = qpm[(size_t)p * 3 + 1], qz = qpm[(size_t)p * 3 + 2];
  int cnt = 0, first = 0; bool found = false;
  for (int j = 0; j < n && cnt < S; ++j) {
    float dx = pm[(size_t)j * 3] - qx, dy = pm[(size_t)j * 3 + 1] - qy, dz = pm[(size_t)j * 3 + 2] - qz;
    float d = dx * dx + dy * dy + dz * dz;
    if (d <= r2) { if (!found) { found = true; first = j; } idx[(size_t)p * S + cnt] = j; ++cnt; }
  }
  for (; cnt < S; ++cnt) idx[(size_t)p * S + cnt] = first;
}

// grouped MLP input rows: [dx dy dz | feat[j][0..Cf) | 0...], row = p*S+s
__global__ void k_group(const float* __restrict__ pm, const float* __restrict__ qpm,
                        const h16* __restrict__ feat, int fKp, int Cf,
                        const int* __restrict__ idx, int P, int S, int dKp,
                        h16* __restrict__ out){
  long long rows = (long long)P * S;
  long long t = (long long)blockIdx.x * blockDim.x + threadIdx.x;
  if (t >= rows) return;
  int p = (int)(t / S);
  int j = idx[t];
  h16* r = out + (size_t)t * dKp;
  r[0] = (h16)(pm[(size_t)j * 3]     - qpm[(size_t)p * 3]);
  r[1] = (h16)(pm[(size_t)j * 3 + 1] - qpm[(size_t)p * 3 + 1]);
  r[2] = (h16)(pm[(size_t)j * 3 + 2] - qpm[(size_t)p * 3 + 2]);
  const h16* fr = feat + (size_t)j * fKp;
  for (int c = 3; c < dKp; ++c) {
    int cf = c - 3;
    r[c] = (cf < Cf) ? fr[cf] : (h16)0.f;
  }
}

// 3-NN with inverse-distance weights
__global__ void k_knn3(const float* __restrict__ pm1, const float* __restrict__ pm2,
                       int n1, int n2, int* __restrict__ i3, float* __restrict__ w3){
  int i = blockIdx.x * blockDim.x + threadIdx.x;
  if (i >= n1) return;
  float x = pm1[(size_t)i * 3], y = pm1[(size_t)i * 3 + 1], z = pm1[(size_t)i * 3 + 2];
  float d0 = 3e38f, d1 = 3e38f, d2 = 3e38f; int j0 = 0, j1 = 0, j2 = 0;
  for (int j = 0; j < n2; ++j) {
    float dx = pm2[(size_t)j * 3] - x, dy = pm2[(size_t)j * 3 + 1] - y, dz = pm2[(size_t)j * 3 + 2] - z;
    float d = dx * dx + dy * dy + dz * dz;
    if (d < d0)      { d2 = d1; j2 = j1; d1 = d0; j1 = j0; d0 = d; j0 = j; }
    else if (d < d1) { d2 = d1; j2 = j1; d1 = d; j1 = j; }
    else if (d < d2) { d2 = d; j2 = j; }
  }
  float r0 = 1.f / (d0 + 1e-8f), r1 = 1.f / (d1 + 1e-8f), r2v = 1.f / (d2 + 1e-8f);
  float s = r0 + r1 + r2v;
  i3[(size_t)i * 3] = j0; i3[(size_t)i * 3 + 1] = j1; i3[(size_t)i * 3 + 2] = j2;
  w3[(size_t)i * 3] = r0 / s; w3[(size_t)i * 3 + 1] = r1 / s; w3[(size_t)i * 3 + 2] = r2v / s;
}

// weighted 3-NN interpolation into dst column range [off, off+C2)
__global__ void k_interp(const h16* __restrict__ f2, int f2Kp,
                         const int* __restrict__ i3, const float* __restrict__ w3,
                         int C2, long long n1, h16* __restrict__ dst, int dKp, int off){
  long long t = (long long)blockIdx.x * blockDim.x + threadIdx.x;
  if (t >= n1 * C2) return;
  long long nrow = t / C2; int c = (int)(t % C2);
  const int*   ii = i3 + (size_t)nrow * 3;
  const float* ww = w3 + (size_t)nrow * 3;
  float v = ww[0] * (float)f2[(size_t)ii[0] * f2Kp + c]
          + ww[1] * (float)f2[(size_t)ii[1] * f2Kp + c]
          + ww[2] * (float)f2[(size_t)ii[2] * f2Kp + c];
  dst[(size_t)nrow * dKp + off + c] = (h16)v;
}

__global__ void k_heat_out(const h16* __restrict__ hT, float* __restrict__ d, int n){
  int i = blockIdx.x * blockDim.x + threadIdx.x;
  if (i < n) d[i] = (float)hT[(size_t)i * 32];
}

// ---------------------------------------------------------------------------
// host-side parameter bookkeeping
// ---------------------------------------------------------------------------
struct ConvP { const float* w; const float* b; };
struct BnP   { const float* g; const float* b; const float* m; const float* v; };
struct MlpP  { ConvP c[3]; BnP bn[3]; };

extern "C" void kernel_launch(void* const* d_in, const int* in_sizes, int n_in,
                              void* d_out, int out_size, void* d_ws, size_t ws_size,
                              hipStream_t stream) {
  (void)in_sizes; (void)n_in; (void)out_size; (void)ws_size;
  const int N = 8192;

  // ---- parameter leaves, in make_params() insertion order (xyz is d_in[0])
  int pi = 1;
  auto getp    = [&]() { return (const float*)d_in[pi++]; };
  auto getconv = [&]() { ConvP c; c.w = getp(); c.b = getp(); return c; };
  auto getbn   = [&]() { BnP b; b.g = getp(); b.b = getp(); b.m = getp(); b.v = getp(); return b; };
  auto getmlp  = [&](int nl) { MlpP m; for (int i = 0; i < nl; ++i) m.c[i] = getconv();
                               for (int i = 0; i < nl; ++i) m.bn[i] = getbn(); return m; };

  ConvP s_c1 = getconv(), s_c2 = getconv(), s_c3 = getconv();
  ConvP s_f1 = getconv(), s_f2 = getconv(), s_f3 = getconv();
  BnP s_b1 = getbn(), s_b2 = getbn(), s_b3 = getbn(), s_b4 = getbn(), s_b5 = getbn();
  ConvP f_c1 = getconv(), f_c2 = getconv(), f_c3 = getconv();
  BnP f_b1 = getbn(), f_b2 = getbn(), f_b3 = getbn();
  ConvP h_c1 = getconv(), h_c2 = getconv(), h_c3 = getconv(), h_c4 = getconv();
  BnP h_b1 = getbn(), h_b2 = getbn(), h_b3 = getbn();
  MlpP sa01 = getmlp(3), sa02 = getmlp(3), sa03 = getmlp(3);
  MlpP fp03 = getmlp(2), fp02 = getmlp(2), fp01 = getmlp(3);
  ConvP conv00 = getconv();
  MlpP sa11 = getmlp(3), sa12 = getmlp(3), sa13 = getmlp(3);
  MlpP fp13 = getmlp(2), fp12 = getmlp(2), fp11 = getmlp(3);
  ConvP conv10 = getconv(), conv11 = getconv(), conv12 = getconv(), conv13 = getconv(), conv14 = getconv();
  (void)sa13; (void)fp13; // dead branch in the reference graph (pruned)

  static const int d_sa01[4] = {20, 64, 64, 128};
  static const int d_sa02[4] = {131, 128, 128, 256};
  static const int d_sa03[4] = {259, 256, 512, 1024};
  static const int d_fp03[3] = {1280, 256, 256};
  static const int d_fp02[3] = {384, 256, 128};
  static const int d_fp01[4] = {134, 128, 128, 128};
  static const int d_sa11[4] = {131, 64, 64, 128};
  static const int d_sa12[4] = {131, 128, 128, 256};
  static const int d_fp12[3] = {384, 256, 128};
  static const int d_fp11[4] = {259, 128, 128, 128};

  // ---- workspace bump allocator with stage marks
  char* wsp = (char*)d_ws;
  size_t off = 0;
  auto alloc = [&](size_t bytes) -> void* {
    void* p = wsp + off;
    off = (off + bytes + 255) & ~(size_t)255;
    return p;
  };

  // ---- building blocks (activations point-major [cols][pad32(C)])
  auto run_cbr = [&](const float* w, const float* bias, const BnP* bn, int cout, int cin,
                     const h16* inT, h16* outT, long long cols, int relu) {
    int Mp = pad32(cout), Kp = pad32(cin);
    long long npk = (long long)(Mp / 16) * (Kp / 32) * 256;
    unsigned* wp = (unsigned*)alloc(4 * npk);
    k_packA<<<gr(npk), TPB, 0, stream>>>(w, wp, cout, cin, Mp, Kp);
    float* sc = (float*)alloc(4ll * Mp);
    float* sh = (float*)alloc(4ll * Mp);
    if (bn) k_fold_bn<<<gr(Mp), TPB, 0, stream>>>(bias, bn->g, bn->b, bn->m, bn->v, sc, sh, cout, Mp);
    else    k_fold_bias<<<gr(Mp), TPB, 0, stream>>>(bias, sc, sh, cout, Mp);
    dim3 grid((unsigned)(cols / 64), (unsigned)(Mp / 32));
    k_gemm<<<grid, TPB, 0, stream>>>((const uint4*)wp, inT, sc, sh, outT, Mp, Kp, relu);
  };

  auto run_fc = [&](ConvP c, const BnP* bn, int M, int K, const h16* x, h16* yh, float* yf, int relu) {
    float* sc = (float*)alloc(4ll * M);
    float* sh = (float*)alloc(4ll * M);
    if (bn) k_fold_bn<<<gr(M), TPB, 0, stream>>>(c.b, bn->g, bn->b, bn->m, bn->v, sc, sh, M, M);
    else    k_fold_bias<<<gr(M), TPB, 0, stream>>>(c.b, sc, sh, M, M);
    k_gemv<<<gr(M), TPB, 0, stream>>>(c.w, x, sc, sh, yh, yf, M, K, relu);
  };

  auto run_mlp = [&](const MlpP& m, const int* dims, int nl, const h16* inT, long long cols,
                     h16* out_last) -> const h16* {
    const h16* cur = inT;
    for (int l = 0; l < nl; ++l) {
      h16* o = (l == nl - 1 && out_last) ? out_last
                                         : (h16*)alloc(2ll * pad32(dims[l + 1]) * cols);
      run_cbr(m.c[l].w, m.c[l].b, &m.bn[l], dims[l + 1], dims[l], cur, o, cols, 1);
      cur = o;
    }
    return cur;
  };

  auto run_sa = [&](const float* pm_in, int n_in, const h16* featT, int fKp, int Cf,
                    const MlpP& mlp, const int* dims, int nl, int P, int S, float r2,
                    int* fidx, bool do_fps, float* newpm, bool do_gather, h16* outT) {
    size_t mk = off;
    if (do_fps) {
      float* dw = (float*)alloc(4ll * n_in);
      k_fps<<<1, 1024, 0, stream>>>(pm_in, n_in, P, fidx, dw);
    }
    if (do_gather) k_gather_xyz<<<gr(P), TPB, 0, stream>>>(pm_in, fidx, P, newpm);
    int* bidx = (int*)alloc(4ll * P * S);
    k_ballquery<<<gr(P), TPB, 0, stream>>>(pm_in, newpm, n_in, P, S, r2, bidx);
    long long rows = (long long)P * S;
    int gKp = pad32(dims[0]);
    h16* gin = (h16*)alloc(2ll * gKp * rows);
    k_group<<<gr(rows), TPB, 0, stream>>>(pm_in, newpm, featT, fKp, Cf, bidx, P, S, gKp, gin);
    const h16* mo = run_mlp(mlp, dims, nl, gin, rows, nullptr);
    int Cp = pad32(dims[nl]);
    k_maxpool<<<gr((long long)P * Cp), TPB, 0, stream>>>(mo, outT, P, S, Cp);
    off = mk;
  };

  auto run_fp = [&](const float* pm1, long long n1, const float* pm2, int n2,
                    const h16* p1a, int aKp, int C1a,
                    const h16* p1b, int bKp, int C1b,
                    const h16* pts2, int p2Kp, int C2,
                    const MlpP& mlp, const int* dims, int nl, h16* outT) {
    size_t mk = off;
    int dKp = pad32(dims[0]);
    h16* cat = (h16*)alloc(2ll * dKp * n1);
    k_cat2<<<gr(n1 * dKp), TPB, 0, stream>>>(cat, dKp, p1a, aKp, C1a, p1b, bKp, C1b, n1);
    int*   i3 = (int*)alloc(12ll * n1);
    float* w3 = (float*)alloc(12ll * n1);
    k_knn3<<<gr(n1), TPB, 0, stream>>>(pm1, pm2, (int)n1, n2, i3, w3);
    k_interp<<<gr(n1 * C2), TPB, 0, stream>>>(pts2, p2Kp, i3, w3, C2, n1, cat, dKp, C1a + C1b);
    run_mlp(mlp, dims, nl, cat, n1, outT);
    off = mk;
  };

  // ---- persistent activations
  const float* xyz = (const float*)d_in[0];
  float* xyz_pm  = (float*)alloc(4ll * N * 3);
  h16*   xyzT    = (h16*)alloc(2ll * N * 32);
  h16*   l0p00T  = (h16*)alloc(2ll * N * 32);            // [xyz(3)|seg(14)|0]
  int*   fps_idx1 = (int*)alloc(4ll * 4096);
  float* nx1_pm  = (float*)alloc(4ll * 4096 * 3);        // l1x0 (prefix 2048 = pipeline-1 l1x)
  h16*   l1p_saT = (h16*)alloc(2ll * 4096 * 128);
  float* nx2_pm  = (float*)alloc(4ll * 2048 * 3);
  h16*   l2p_saT = (h16*)alloc(2ll * 2048 * 256);
  h16*   l3vec   = (h16*)alloc(2ll * 1024);
  h16*   l2p0T   = (h16*)alloc(2ll * 2048 * 256);        // fp03 out (feeds fp02 and fp12)
  h16*   l1p0T   = (h16*)alloc(2ll * 4096 * 128);
  h16*   l0p0T   = (h16*)alloc(2ll * N * 128);
  h16*   l0p1T   = (h16*)alloc(2ll * N * 128);
  h16*   l1p_bT  = (h16*)alloc(2ll * 2048 * 128);
  float* nx12_pm = (float*)alloc(4ll * 1024 * 3);
  h16*   l2p_bT  = (h16*)alloc(2ll * 1024 * 256);
  h16*   l1p_b2T = (h16*)alloc(2ll * 2048 * 128);
  h16*   l0p_bT  = (h16*)alloc(2ll * N * 128);
  float* trans9  = (float*)alloc(64);

  k_prep_xyz<<<gr(N), TPB, 0, stream>>>(xyz, xyz_pm, xyzT, N);

  // ================= STN + feat + head (transient region) =================
  {
    size_t mk = off;
    h16* t1 = (h16*)alloc(2ll * N * 64);
    h16* t2 = (h16*)alloc(2ll * N * 128);
    h16* t3 = (h16*)alloc(2ll * N * 1024);
    run_cbr(s_c1.w, s_c1.b, &s_b1, 64, 3, xyzT, t1, N, 1);
    run_cbr(s_c2.w, s_c2.b, &s_b2, 128, 64, t1, t2, N, 1);
    run_cbr(s_c3.w, s_c3.b, &s_b3, 1024, 128, t2, t3, N, 1);
    h16* tmax = (h16*)alloc(2ll * 1024);
    k_colmax<<<gr(1024), TPB, 0, stream>>>(t3, tmax, N, 1024);
    h16* v1 = (h16*)alloc(2ll * 512);
    h16* v2 = (h16*)alloc(2ll * 256);
    run_fc(s_f1, &s_b4, 512, 1024, tmax, v1, nullptr, 1);
    run_fc(s_f2, &s_b5, 256, 512, v1, v2, nullptr, 1);
    run_fc(s_f3, nullptr, 9, 256, v2, nullptr, trans9, 0);
    k_add_eye<<<1, 32, 0, stream>>>(trans9);

    h16* e1 = (h16*)alloc(2ll * N * 32);
    k_transform<<<gr(N), TPB, 0, stream>>>(xyz, trans9, e1, N);
    h16* pf = (h16*)alloc(2ll * N * 64);
    h16* e2 = (h16*)alloc(2ll * N * 128);
    h16* e3 = (h16*)alloc(2ll * N * 1024);
    run_cbr(f_c1.w, f_c1.b, &f_b1, 64, 3, e1, pf, N, 1);
    run_cbr(f_c2.w, f_c2.b, &f_b2, 128, 64, pf, e2, N, 1);
    run_cbr(f_c3.w, f_c3.b, &f_b3, 1024, 128, e2, e3, N, 0);   // bn, no relu
    h16* gvec = (h16*)alloc(2ll * 1024);
    k_colmax<<<gr(1024), TPB, 0, stream>>>(e3, gvec, N, 1024);

    h16* featcat = (h16*)alloc(2ll * N * 1088);
    k_vec_cat<<<gr((long long)N * 1088), TPB, 0, stream>>>(featcat, 1088, gvec, 1024, pf, 64, 64, N);

    h16* hx1 = (h16*)alloc(2ll * N * 512);
    h16* hx2 = (h16*)alloc(2ll * N * 256);
    h16* hx3 = (h16*)alloc(2ll * N * 128);
    h16* logits = (h16*)alloc(2ll * N * 32);
    run_cbr(h_c1.w, h_c1.b, &h_b1, 512, 1088, featcat, hx1, N, 1);
    run_cbr(h_c2.w, h_c2.b, &h_b2, 256, 512, hx1, hx2, N, 1);
    run_cbr(h_c3.w, h_c3.b, &h_b3, 128, 256, hx2, hx3, N, 1);
    run_cbr(h_c4.w, h_c4.b, nullptr, 14, 128, hx3, logits, N, 0);
    k_logsm14<<<gr(N), TPB, 0, stream>>>(logits, xyzT, (float*)d_out, l0p00T, N);
    off = mk;
  }

  // ================= pipeline 0 =================
  run_sa(xyz_pm, N, l0p00T, 32, 17, sa01, d_sa01, 3, 4096, 256, 0.04f,
         fps_idx1, true, nx1_pm, true, l1p_saT);
  {
    size_t mk = off;
    int* fidx2 = (int*)alloc(4ll * 2048);
    run_sa(nx1_pm, 4096, l1p_saT, 128, 128, sa02, d_sa02, 3, 2048, 256, 0.16f,
           fidx2, true, nx2_pm, true, l2p_saT);
    off = mk;
  }
  { // sa03 (group_all): [xyz(3)|l2p(256)|0] over 2048 rows, MLP, global max
    size_t mk = off;
    int cKp = pad32(259);
    h16* cat = (h16*)alloc(2ll * 2048 * cKp);
    k_pm_cat<<<gr(2048ll * cKp), TPB, 0, stream>>>(cat, cKp, nx2_pm, l2p_saT, 256, 256, 2048);
    const h16* mo = run_mlp(sa03, d_sa03, 3, cat, 2048, nullptr);
    k_colmax<<<gr(1024), TPB, 0, stream>>>(mo, l3vec, 2048, 1024);
    off = mk;
  }
  { // fp03 (s==1 -> broadcast interp): [l2p(256)|l3vec(1024)]
    size_t mk = off;
    h16* cat = (h16*)alloc(2ll * 2048 * 1280);
    k_featvec_cat<<<gr(2048ll * 1280), TPB, 0, stream>>>(cat, 1280, l2p_saT, 256, 256, l3vec, 1024, 2048);
    run_mlp(fp03, d_fp03, 2, cat, 2048, l2p0T);
    off = mk;
  }
  run_fp(nx1_pm, 4096, nx2_pm, 2048, l1p_saT, 128, 128, nullptr, 0, 0,
         l2p0T, 256, 256, fp02, d_fp02, 2, l1p0T);
  run_fp(xyz_pm, N, nx1_pm, 4096, xyzT, 32, 3, xyzT, 32, 3,
         l1p0T, 128, 128, fp01, d_fp01, 3, l0p0T);
  run_cbr(conv00.w, conv00.b, nullptr, 128, 128, l0p0T, l0p1T, N, 0);

  // ================= pipeline 1 =================
  run_sa(xyz_pm, N, l0p1T, 128, 128, sa11, d_sa11, 3, 2048, 256, 0.04f,
         fps_idx1, false, nx1_pm, false, l1p_bT);   // reuse FPS prefix + centroids
  {
    size_t mk = off;
    int* fidx12 = (int*)alloc(4ll * 1024);
    run_sa(nx1_pm, 2048, l1p_bT, 128, 128, sa12, d_sa12, 3, 1024, 256, 0.16f,
           fidx12, true, nx12_pm, true, l2p_bT);
    off = mk;
  }
  run_fp(nx1_pm, 2048, nx12_pm, 1024, l1p_bT, 128, 128, nullptr, 0, 0,
         l2p0T, 256, 256, fp12, d_fp12, 2, l1p_b2T);
  run_fp(xyz_pm, N, nx1_pm, 2048, xyzT, 32, 3, l0p1T, 128, 128,
         l1p_b2T, 128, 128, fp11, d_fp11, 3, l0p_bT);

  // ================= heatmap head =================
  {
    size_t mk = off;
    h16* net = (h16*)alloc(2ll * N * 512);
    run_cbr(conv10.w, conv10.b, nullptr, 512, 128, l0p_bT, net, N, 0);
    int c1Kp = pad32(515);
    h16* cat1 = (h16*)alloc(2ll * N * c1Kp);
    k_cat2<<<gr((long long)N * c1Kp), TPB, 0, stream>>>(cat1, c1Kp, xyzT, 32, 3, net, 512, 512, N);
    h16* net2 = (h16*)alloc(2ll * N * 128);
    run_cbr(conv11.w, conv11.b, nullptr, 128, 515, cat1, net2, N, 0);
    int c2Kp = pad32(131);
    h16* cat2 = (h16*)alloc(2ll * N * c2Kp);
    k_cat2<<<gr((long long)N * c2Kp), TPB, 0, stream>>>(cat2, c2Kp, xyzT, 32, 3, net2, 128, 128, N);
    h16* net3 = (h16*)alloc(2ll * N * 32);
    run_cbr(conv12.w, conv12.b, nullptr, 32, 131, cat2, net3, N, 0);
    int c3Kp = pad32(35);
    h16* cat3 = (h16*)alloc(2ll * N * c3Kp);
    k_cat2<<<gr((long long)N * c3Kp), TPB, 0, stream>>>(cat3, c3Kp, xyzT, 32, 3, net3, 32, 32, N);
    h16* net4 = (h16*)alloc(2ll * N * 32);
    run_cbr(conv13.w, conv13.b, nullptr, 1, 35, cat3, net4, N, 0);
    h16* heatT = (h16*)alloc(2ll * N * 32);
    run_cbr(conv14.w, conv14.b, nullptr, 1, 1, net4, heatT, N, 0);
    k_heat_out<<<gr(N), TPB, 0, stream>>>(heatT, (float*)d_out + 14ll * N, N);
    off = mk;
  }
}
